// ContMixT_58076547776817
// MI455X (gfx1250) — compile-verified, matrix operand order checked by reading, not run on hardware
//
#include <hip/hip_runtime.h>
#include <math.h>
#include <stdint.h>

typedef __attribute__((ext_vector_type(16))) _Float16 v16h;
typedef __attribute__((ext_vector_type(8)))  float    v8f;

// ---------------------------------------------------------------- constants
constexpr int BN  = 8;
constexpr int CC  = 256;
constexpr int HH  = 48;
constexpr int WW  = 80;
constexpr int HWp = HH * WW;          // 3840
constexpr int MT  = BN * HWp;         // 30720
constexpr int TM  = 128;              // block tile M
constexpr int TN  = 64;               // block tile N
constexpr int KST = 32;               // K step (WMMA f16 K)
constexpr int ASTR = 40;              // LDS A row stride (halves, 80B -> b128 aligned)

// ================================================================ prep: f32 [b][c][s] -> f16 spatial-major [b][s][cc] (tiled transpose)
__global__ __launch_bounds__(256) void pack_x_kernel(
    const float* __restrict__ a, const float* __restrict__ b,
    const float* __restrict__ c, _Float16* __restrict__ X)
{
  // grid = 8 * 24 * 120 ; tile = 32 channels x 32 spatial
  int blk = blockIdx.x;
  int sg = blk % 120;
  int cg = (blk / 120) % 24;
  int bb = blk / (120 * 24);
  const float* src = (cg < 8) ? a : (cg < 16) ? b : c;
  int c0 = (cg & 7) * 32;
  int s0 = sg * 32;
  __shared__ float tile[32][33];
  int t = threadIdx.x;
  int i = t >> 3, j0 = (t & 7) * 4;
  const float* sp = src + ((size_t)bb * 256 + c0 + i) * HWp + s0 + j0;
  tile[i][j0 + 0] = sp[0]; tile[i][j0 + 1] = sp[1];
  tile[i][j0 + 2] = sp[2]; tile[i][j0 + 3] = sp[3];
  __syncthreads();
  _Float16* xp = X + ((size_t)bb * HWp + s0 + i) * 768 + cg * 32 + j0;
  xp[0] = (_Float16)tile[j0 + 0][i]; xp[1] = (_Float16)tile[j0 + 1][i];
  xp[2] = (_Float16)tile[j0 + 2][i]; xp[3] = (_Float16)tile[j0 + 3][i];
}

// ================================================================ prep: weights -> per-lane WMMA B-fragment layout
// K ordering: k = tap*CIN + c   (tap = ky*3+kx)
__global__ __launch_bounds__(256) void pack_w_kernel(
    const float* __restrict__ Wsrc, _Float16* __restrict__ Wp, int Cin, int Ksub)
{
  int id   = blockIdx.x * 256 + threadIdx.x;
  int e    = id & 15;
  int lane = (id >> 4) & 31;
  int rest = id >> 9;
  int ks   = rest % Ksub;
  int ns   = rest / Ksub;
  int v = e >> 1, j = e & 1;
  int n  = (ns << 4) + (lane & 15);
  int kk = ((v >> 2) << 4) + ((lane >> 4) << 3) + ((v & 3) << 1) + j;
  int k  = ks * KST + kk;
  int r  = k / Cin;                    // tap
  int c  = k - r * Cin;                // input channel
  Wp[id] = (_Float16)Wsrc[((size_t)n * Cin + c) * 9 + r];
}

// ================================================================ prep: fold conv bias + BN into per-channel scale/shift
__global__ void bnprep_kernel(
    const float* cb1, const float* g1, const float* b1, const float* m1, const float* v1,
    const float* cb2, const float* g2, const float* b2, const float* m2, const float* v2,
    float* A1, float* B1, float* A2, float* B2)
{
  int t = threadIdx.x;
  float s1 = g1[t] * rsqrtf(v1[t] + 1e-5f);
  A1[t] = s1;  B1[t] = (cb1[t] - m1[t]) * s1 + b1[t];
  float s2 = g2[t] * rsqrtf(v2[t] + 1e-5f);
  A2[t] = s2;  B2[t] = (cb2[t] - m2[t]) * s2 + b2[t];
}

// ================================================================ WMMA implicit-GEMM dilated 3x3 conv
// X spatial-major f16 [b][s][CIN] ; Wp packed B fragments
// Double-buffered LDS, async global->LDS staging pipelined with WMMA,
// single barrier per K-step.
// Hout != null -> conv1: f16 [m][256] out ; else conv2: pooled accumulation
template<int CIN, int DIL>
__global__ __launch_bounds__(256) void conv_gemm_kernel(
    const _Float16* __restrict__ X, const _Float16* __restrict__ Wp,
    const float* __restrict__ bnA, const float* __restrict__ bnB,
    _Float16* __restrict__ Hout, float* __restrict__ pooled)
{
  constexpr int NK2  = CIN / KST;      // K-steps per tap
  constexpr int NKS  = 9 * NK2;        // total K-steps
  constexpr int ABUF = TM * ASTR;      // halves per LDS buffer

  const int t    = threadIdx.x;
  const int lane = t & 31;
  const int wv   = t >> 5;
  const int wm   = wv & 3;             // wave M group (4)
  const int wn   = wv >> 2;            // wave N group (2)
  const int l15  = lane & 15;
  const int h16  = lane >> 4;

  const int mBase = blockIdx.x * TM;
  const int nTile = blockIdx.y;        // 0..3
  const int bImg  = mBase / HWp;

  __shared__ __align__(16) _Float16 As[2 * ABUF];   // 20480 B

  // this thread's A staging slice: row mmL, 16 halves starting at kk0
  const int mmL = t & 127;
  const int kk0 = (t >> 7) * 16;
  const int mG  = mBase + mmL;
  const int sG  = mG % HWp;
  const int yG  = sG / WW;
  const int xG  = sG - yG * WW;
  const unsigned ldsA0 = (unsigned)(uintptr_t)&As[mmL * ASTR + kk0];

  // ---- incremental staging state (tap geometry recomputed 9x total)
  int tapS = 0, ks2S = 0;
  bool validS = false;
  unsigned long long gaS = 0;
  auto setTap = [&](int tap) {
    int ky = tap / 3, kx = tap - (tap / 3) * 3;
    int yy = yG + (ky - 1) * DIL;
    int xx = xG + (kx - 1) * DIL;
    validS = ((unsigned)yy < (unsigned)HH) && ((unsigned)xx < (unsigned)WW);
    gaS = (unsigned long long)(uintptr_t)(X + ((size_t)bImg * HWp + yy * WW + xx) * CIN + kk0);
  };
  setTap(0);

  auto stage = [&](int buf) {
    if (validS) {
      unsigned ld = ldsA0 + (unsigned)(buf * ABUF * 2);
      asm volatile("global_load_async_to_lds_b128 %0, %1, off"
                   :: "v"(ld), "v"(gaS) : "memory");
      asm volatile("global_load_async_to_lds_b128 %0, %1, off offset:16"
                   :: "v"(ld), "v"(gaS) : "memory");
    } else {
      v16h z = {};
      *(v16h*)&As[buf * ABUF + mmL * ASTR + kk0] = z;
    }
    gaS += (unsigned long long)(KST * 2);           // 64 B per K-step
    if (++ks2S == NK2) { ks2S = 0; ++tapS; if (tapS < 9) setTap(tapS); }
  };

  stage(0);                                          // prologue: tile 0 -> buf 0

  v8f acc[2][2] = {};
  const int m0  = wm * 32 + l15;                     // A fragment row base
  const int ns0 = nTile * 4 + wn * 2;

  for (int ks = 0; ks < NKS; ++ks) {
    // tile ks was staged one iteration ago (or in the prologue)
    asm volatile("s_wait_asynccnt 0x0" ::: "memory");
    __syncthreads();

    // stage tile ks+1 (async, lands while we do WMMA below)
    if (ks + 1 < NKS) stage((ks + 1) & 1);

    // ---- B fragments straight from pre-swizzled global (coalesced b128)
    const v16h bf0 = *(const v16h*)(Wp + (((size_t)ns0       * NKS + ks) * 32 + lane) * 16);
    const v16h bf1 = *(const v16h*)(Wp + (((size_t)(ns0 + 1) * NKS + ks) * 32 + lane) * 16);
    if (ks + 1 < NKS)
      __builtin_prefetch(Wp + (((size_t)ns0 * NKS + ks + 1) * 32 + lane) * 16, 0, 1);

    // ---- A fragments from LDS (documented 16-bit A layout)
    const _Float16* Ab = As + (ks & 1) * ABUF;
    union AF { v16h v; unsigned u[8]; };
    AF a0, a1;
#pragma unroll
    for (int v = 0; v < 8; ++v) {
      int kkv = ((v >> 2) << 4) + (h16 << 3) + ((v & 3) << 1);
      a0.u[v] = *(const unsigned*)&Ab[(m0)      * ASTR + kkv];
      a1.u[v] = *(const unsigned*)&Ab[(m0 + 16) * ASTR + kkv];
    }

    acc[0][0] = __builtin_amdgcn_wmma_f32_16x16x32_f16(false, a0.v, false, bf0, (short)0, acc[0][0], false, false);
    acc[0][1] = __builtin_amdgcn_wmma_f32_16x16x32_f16(false, a0.v, false, bf1, (short)0, acc[0][1], false, false);
    acc[1][0] = __builtin_amdgcn_wmma_f32_16x16x32_f16(false, a1.v, false, bf0, (short)0, acc[1][0], false, false);
    acc[1][1] = __builtin_amdgcn_wmma_f32_16x16x32_f16(false, a1.v, false, bf1, (short)0, acc[1][1], false, false);
  }

  // ---- fused epilogue: y = relu(acc*A + B)
  if (Hout) {
    // conv1: direct register -> global f16 store, [m][c] layout (lanes = consecutive c)
#pragma unroll
    for (int mi = 0; mi < 2; ++mi)
#pragma unroll
      for (int ni = 0; ni < 2; ++ni) {
        int nG = nTile * 64 + wn * 32 + ni * 16 + l15;
        float sA = bnA[nG], sB = bnB[nG];
#pragma unroll
        for (int v = 0; v < 8; ++v) {
          int m = mBase + wm * 32 + mi * 16 + v + 8 * h16;
          float y = fmaf(acc[mi][ni][v], sA, sB);
          y = y > 0.0f ? y : 0.0f;
          Hout[(size_t)m * CC + nG] = (_Float16)y;
        }
      }
  } else {
    // conv2: relu + spatial sum for global-average-pool (h2 never materialized)
#pragma unroll
    for (int ni = 0; ni < 2; ++ni) {
      int nG = nTile * 64 + wn * 32 + ni * 16 + l15;
      float sA = bnA[nG], sB = bnB[nG];
      float part = 0.0f;
#pragma unroll
      for (int mi = 0; mi < 2; ++mi)
#pragma unroll
        for (int v = 0; v < 8; ++v) {
          float y = fmaf(acc[mi][ni][v], sA, sB);
          part += (y > 0.0f ? y : 0.0f);
        }
      part += __shfl_down(part, 16, 32);
      if (h16 == 0)
        atomicAdd(pooled + (size_t)bImg * CC + nG, part);
    }
  }
}

// ================================================================ G = 1x1 conv of pooled mean
__global__ void g_kernel(const float* __restrict__ gw, const float* __restrict__ gb,
                         const float* __restrict__ pooled, float* __restrict__ G)
{
  int b = blockIdx.x, c = threadIdx.x;
  float s = gb[c];
  const float inv = 1.0f / (float)HWp;
  for (int hd = 0; hd < 256; ++hd)
    s += gw[c * 256 + hd] * pooled[b * 256 + hd] * inv;
  G[b * 256 + c] = s;
}

// ================================================================ gap over Fs = [f_t ; f_t-1]
__global__ void gap_kernel(const float* __restrict__ f_t, const float* __restrict__ f_tm1,
                           float* __restrict__ gap)
{
  int n = blockIdx.x >> 8, c = blockIdx.x & 255;
  const float* src = (n < 8) ? f_t   + ((size_t)(n    ) * 256 + c) * HWp
                             : f_tm1 + ((size_t)(n - 8) * 256 + c) * HWp;
  float s = 0.0f;
  for (int i = threadIdx.x; i < HWp; i += 256) s += src[i];
  __shared__ float red[256];
  red[threadIdx.x] = s; __syncthreads();
  for (int off = 128; off; off >>= 1) {
    if (threadIdx.x < off) red[threadIdx.x] += red[threadIdx.x + off];
    __syncthreads();
  }
  if (threadIdx.x == 0) gap[blockIdx.x] = red[0] * (1.0f / (float)HWp);
}

// ================================================================ fc1 (silu)
__global__ void fc1_kernel(const float* __restrict__ G, const float* __restrict__ gap,
                           const float* __restrict__ w, const float* __restrict__ bias,
                           float* __restrict__ y1)
{
  int n = blockIdx.x, t = threadIdx.x;
  __shared__ float fin[512];
  fin[t]       = G[(n >> 1) * 256 + t];      // g_flat = repeat(G, 2) -> row n//2
  fin[256 + t] = gap[n * 256 + t];
  __syncthreads();
  for (int j = t; j < 512; j += 256) {
    float s = bias[j];
    for (int i = 0; i < 512; ++i) s += fin[i] * w[j * 512 + i];
    y1[n * 512 + j] = s / (1.0f + expf(-s));
  }
}

// ================================================================ fc2 -> dynamic depthwise weights
__global__ void fc2_kernel(const float* __restrict__ y1, const float* __restrict__ w,
                           const float* __restrict__ bias, float* __restrict__ wd)
{
  int n = blockIdx.y, t = threadIdx.x;
  __shared__ float yr[512];
  yr[t] = y1[n * 512 + t]; yr[256 + t] = y1[n * 512 + 256 + t];
  __syncthreads();
  int o = blockIdx.x * 256 + t;
  float s = bias[o];
  for (int i = 0; i < 512; ++i) s += yr[i] * w[o * 512 + i];
  wd[n * 2304 + o] = s;
}

// ================================================================ dynamic depthwise 3x3 (groups = 2*B*C)
__global__ void dw_kernel(const float* __restrict__ f_t, const float* __restrict__ f_tm1,
                          const float* __restrict__ wd,
                          float* __restrict__ ftmod, float* __restrict__ f1mod)
{
  int plane = blockIdx.x;                   // 0..4095 = n*256 + c
  int n = plane >> 8, c = plane & 255, t = threadIdx.x;
  const float* src = (n < 8) ? f_t   + ((size_t)(n    ) * 256 + c) * HWp
                             : f_tm1 + ((size_t)(n - 8) * 256 + c) * HWp;
  float* dst       = (n < 8) ? ftmod + ((size_t)(n    ) * 256 + c) * HWp
                             : f1mod + ((size_t)(n - 8) * 256 + c) * HWp;
  __shared__ float w9[9];
  if (t < 9) w9[t] = wd[(size_t)plane * 9 + t];
  __syncthreads();
  for (int p = t; p < HWp; p += 256) {
    int y = p / WW, x = p - y * WW;
    float s = 0.0f;
#pragma unroll
    for (int ky = 0; ky < 3; ++ky) {
      int yy = y + ky - 1;
      if ((unsigned)yy < (unsigned)HH) {
#pragma unroll
        for (int kx = 0; kx < 3; ++kx) {
          int xx = x + kx - 1;
          if ((unsigned)xx < (unsigned)WW) s += w9[ky * 3 + kx] * src[yy * WW + xx];
        }
      }
    }
    dst[p] = s;
  }
}

// ================================================================ fused attention 1x1 convs + alpha blends
__global__ void blend_kernel(const float* __restrict__ ftmod, const float* __restrict__ f1mod,
                             const float* __restrict__ ftm2,
                             const float* __restrict__ apw, const float* __restrict__ apb,
                             const float* __restrict__ aw,  const float* __restrict__ ab,
                             float* __restrict__ out)
{
  int p = blockIdx.x * 256 + threadIdx.x;       // pixel id, blocks never cross images
  int b = p / HWp, s = p - b * HWp;
  size_t base = (size_t)b * 256 * HWp + s;

  float accp = apb[0];
  for (int c = 0; c < 256; ++c) {
    size_t i = base + (size_t)c * HWp;
    accp += apw[c] * f1mod[i] + apw[256 + c] * ftm2[i];
  }
  float al_p = 0.3f + 0.4f / (1.0f + expf(-accp));

  float acca = ab[0];
  for (int c = 0; c < 256; ++c) {
    size_t i = base + (size_t)c * HWp;
    float fp = al_p * f1mod[i] + (1.0f - al_p) * ftm2[i];
    acca += aw[c] * ftmod[i] + aw[256 + c] * fp;
  }
  float al = 0.6f + 0.3f / (1.0f + expf(-acca));

  for (int c = 0; c < 256; ++c) {
    size_t i = base + (size_t)c * HWp;
    float fp = al_p * f1mod[i] + (1.0f - al_p) * ftm2[i];
    out[i] = al * ftmod[i] + (1.0f - al) * fp;
  }
}

// ================================================================ launcher
extern "C" void kernel_launch(void* const* d_in, const int* in_sizes, int n_in,
                              void* d_out, int out_size, void* d_ws, size_t ws_size,
                              hipStream_t stream)
{
  const float* f_tm2 = (const float*)d_in[0];
  const float* f_tm1 = (const float*)d_in[1];
  const float* f_t   = (const float*)d_in[2];
  const float* ov_w1 = (const float*)d_in[3];
  const float* ov_b1 = (const float*)d_in[4];
  const float* bn1_g = (const float*)d_in[5];
  const float* bn1_b = (const float*)d_in[6];
  const float* bn1_m = (const float*)d_in[7];
  const float* bn1_v = (const float*)d_in[8];
  const float* ov_w2 = (const float*)d_in[9];
  const float* ov_b2 = (const float*)d_in[10];
  const float* bn2_g = (const float*)d_in[11];
  const float* bn2_b = (const float*)d_in[12];
  const float* bn2_m = (const float*)d_in[13];
  const float* bn2_v = (const float*)d_in[14];
  const float* g_w   = (const float*)d_in[15];
  const float* g_b   = (const float*)d_in[16];
  const float* fc1_w = (const float*)d_in[17];
  const float* fc1_b = (const float*)d_in[18];
  const float* fc2_w = (const float*)d_in[19];
  const float* fc2_b = (const float*)d_in[20];
  const float* ap_w  = (const float*)d_in[21];
  const float* ap_b  = (const float*)d_in[22];
  const float* a_w   = (const float*)d_in[23];
  const float* a_b   = (const float*)d_in[24];
  float* out = (float*)d_out;

  // ---- workspace layout (all 256B aligned)
  char* ws = (char*)d_ws;
  size_t off = 0;
  auto take = [&](size_t bytes) { char* p = ws + off; off = (off + bytes + 255) & ~(size_t)255; return p; };
  _Float16* xcat = (_Float16*)take((size_t)BN * 768 * HWp * 2);   // 47.2 MB [b][s][768]
  _Float16* w1p  = (_Float16*)take((size_t)6912 * 256 * 2);       //  3.5 MB
  _Float16* h1   = (_Float16*)take((size_t)BN * 256 * HWp * 2);   // 15.7 MB [m][256]
  _Float16* w2p  = (_Float16*)take((size_t)2304 * 256 * 2);       //  1.2 MB
  float* pooled  = (float*)take((size_t)BN * 256 * 4);
  float* Gbuf    = (float*)take((size_t)BN * 256 * 4);
  float* gapb    = (float*)take((size_t)16 * 256 * 4);
  float* y1      = (float*)take((size_t)16 * 512 * 4);
  float* wdyn    = (float*)take((size_t)16 * 2304 * 4);
  float* A1 = (float*)take(1024); float* B1 = (float*)take(1024);
  float* A2 = (float*)take(1024); float* B2 = (float*)take(1024);
  float* ftmod = (float*)take((size_t)BN * 256 * HWp * 4);        // 31.5 MB
  float* f1mod = (float*)take((size_t)BN * 256 * HWp * 4);        // 31.5 MB
  (void)ws_size; (void)in_sizes; (void)n_in; (void)out_size;

  // ---- prep
  pack_x_kernel<<<8 * 24 * 120, 256, 0, stream>>>(f_tm2, f_tm1, f_t, xcat);
  pack_w_kernel<<<(16 * 216 * 512) / 256, 256, 0, stream>>>(ov_w1, w1p, 768, 216);
  pack_w_kernel<<<(16 * 72 * 512) / 256, 256, 0, stream>>>(ov_w2, w2p, 256, 72);
  bnprep_kernel<<<1, 256, 0, stream>>>(ov_b1, bn1_g, bn1_b, bn1_m, bn1_v,
                                       ov_b2, bn2_g, bn2_b, bn2_m, bn2_v,
                                       A1, B1, A2, B2);

  // ---- conv1 (WMMA, fused BN+ReLU, f16 out)
  conv_gemm_kernel<768, 2><<<dim3(MT / TM, 256 / TN), 256, 0, stream>>>(
      xcat, w1p, A1, B1, h1, nullptr);

  // ---- conv2 (WMMA, fused BN+ReLU+global-average-pool)
  hipMemsetAsync(pooled, 0, (size_t)BN * 256 * 4, stream);
  conv_gemm_kernel<256, 4><<<dim3(MT / TM, 256 / TN), 256, 0, stream>>>(
      h1, w2p, A2, B2, nullptr, pooled);

  // ---- small dense tail
  g_kernel  <<<BN, 256, 0, stream>>>(g_w, g_b, pooled, Gbuf);
  gap_kernel<<<16 * 256, 256, 0, stream>>>(f_t, f_tm1, gapb);
  fc1_kernel<<<16, 256, 0, stream>>>(Gbuf, gapb, fc1_w, fc1_b, y1);
  fc2_kernel<<<dim3(9, 16), 256, 0, stream>>>(y1, fc2_w, fc2_b, wdyn);

  // ---- dynamic depthwise + fused blends
  dw_kernel   <<<16 * 256, 256, 0, stream>>>(f_t, f_tm1, wdyn, ftmod, f1mod);
  blend_kernel<<<MT / 256, 256, 0, stream>>>(ftmod, f1mod, f_tm2,
                                             ap_w, ap_b, a_w, a_b, out);
}
